// NeighborEncoder_63866163692317
// MI455X (gfx1250) — matrix operand order repaired and checked by reading, not compiled, not run on hardware
//
#include <hip/hip_runtime.h>
#include <hip/hip_bf16.h>
#include <math.h>

typedef __attribute__((ext_vector_type(16))) __bf16 v16bf;
typedef __attribute__((ext_vector_type(8)))  float  v8f;
typedef unsigned short u16;
typedef unsigned int   u32;

#define D512 512
#define S25  25
#define NH   16
#define HD32 32

// ---------- helpers ----------
union BF16x16 { v16bf v; uint4 q[2]; };

static __device__ __forceinline__ u16 f2bf(float f) {
    union { float f; u32 u; } a; a.f = f;
    u32 r = (a.u + 0x7FFFu + ((a.u >> 16) & 1u)) >> 16;
    return (u16)r;
}

static __device__ __forceinline__ v8f zero8() {
    v8f z = {0.f,0.f,0.f,0.f,0.f,0.f,0.f,0.f};
    return z;
}

// CDNA5 async copy: global -> LDS, 16B per lane, tracked by ASYNCcnt.
static __device__ __forceinline__ void async_g2l_b128(u32 lds_off, const void* gaddr) {
    asm volatile("global_load_async_to_lds_b128 %0, %1, off"
                 :: "v"(lds_off), "v"((unsigned long long)(uintptr_t)gaddr)
                 : "memory");
}
static __device__ __forceinline__ void wait_async0() {
    asm volatile("s_wait_asynccnt 0x0" ::: "memory");
}

// A fragment: 16-bit A matrix 16x32 (M x K), row-major source, stride in elems.
// lane<16: row m0+lane, K chunks [0..7],[16..23]; lane>=16: row m0+lane-16, [8..15],[24..31]
static __device__ __forceinline__ v16bf load_a_frag(const u16* base, int stride, int m0, int k0) {
    int lane = threadIdx.x & 31;
    const u16* p = base + (m0 + (lane & 15)) * stride + k0 + ((lane & 16) ? 8 : 0);
    BF16x16 r;
    r.q[0] = *(const uint4*)(p);
    r.q[1] = *(const uint4*)(p + 16);
    return r.v;
}

// B fragment: 16-bit B matrix 32x16 (K x N). Source is row-major [N][K] (i.e. W s.t.
// out = A @ W^T). lane<16: row n0+lane K chunks [0..7],[16..23]; lane>=16: [8..15],[24..31].
static __device__ __forceinline__ v16bf load_b_frag(const u16* base, int stride, int n0, int k0) {
    int lane = threadIdx.x & 31;
    const u16* p = base + (n0 + (lane & 15)) * stride + k0 + ((lane & 16) ? 8 : 0);
    BF16x16 r;
    r.q[0] = *(const uint4*)(p);
    r.q[1] = *(const uint4*)(p + 16);
    return r.v;
}

static __device__ __forceinline__ v8f wmma_bf16(v16bf a, v16bf b, v8f c) {
    return __builtin_amdgcn_wmma_f32_16x16x32_bf16(false, a, false, b, (short)0, c, false, false);
}

static __device__ __forceinline__ float gelu_exact(float v) {
    return 0.5f * v * (1.0f + erff(v * 0.70710678118654752f));
}

// LayerNorm rows 0..24 of xs -> bf16 dst[32][512], pad rows 25..31 zeroed.
static __device__ __forceinline__ void ln_to_bf16(const float* xs, const float* g,
                                                  const float* bvec, u16* dst) {
    int tid = threadIdx.x, wave = tid >> 5, lane = tid & 31;
    for (int t = wave; t < S25; t += 8) {
        const float4* rp = (const float4*)(xs + t * D512 + lane * 16);
        float4 v[4];
        float s = 0.f, ss = 0.f;
#pragma unroll
        for (int i = 0; i < 4; ++i) {
            v[i] = rp[i];
            s  += v[i].x + v[i].y + v[i].z + v[i].w;
            ss += v[i].x*v[i].x + v[i].y*v[i].y + v[i].z*v[i].z + v[i].w*v[i].w;
        }
#pragma unroll
        for (int off = 16; off; off >>= 1) {
            s  += __shfl_xor(s, off, 32);
            ss += __shfl_xor(ss, off, 32);
        }
        float mu  = s  * (1.f / 512.f);
        float var = ss * (1.f / 512.f) - mu * mu;
        float r   = rsqrtf(var + 1e-5f);
        u16* drow = dst + t * D512 + lane * 16;
#pragma unroll
        for (int i = 0; i < 4; ++i) {
            float vv[4] = { v[i].x, v[i].y, v[i].z, v[i].w };
#pragma unroll
            for (int j = 0; j < 4; ++j) {
                int col = lane * 16 + i * 4 + j;
                drow[i * 4 + j] = f2bf((vv[j] - mu) * r * g[col] + bvec[col]);
            }
        }
    }
    for (int i = tid; i < 7 * D512; i += 256) dst[S25 * D512 + i] = 0;
}

// GEMM: out32x512 = A(bf16 LDS [32][512]) @ W^T(bf16 global [512][512]) + bias
// GELU=false: residual add into xs (rows<25). GELU=true: gelu -> hout bf16 [32][512].
template<bool GELU>
static __device__ __forceinline__ void gemm_ffn(const u16* A, const u16* W, const float* bias,
                                                float* xres, u16* hout) {
    int tid = threadIdx.x, wave = tid >> 5, lane = tid & 31;
    for (int g = wave; g < 16; g += 8) {      // 2 m-tiles x 8 groups of 4 n-tiles
        int mt = g >> 3, ng = g & 7;
        int m0 = mt * 16, n0 = ng * 64;
        v8f acc[4] = { zero8(), zero8(), zero8(), zero8() };
        for (int kt = 0; kt < 16; ++kt) {
            v16bf af = load_a_frag(A, D512, m0, kt * 32);
#pragma unroll
            for (int j = 0; j < 4; ++j) {
                v16bf bf = load_b_frag(W, D512, n0 + j * 16, kt * 32);
                acc[j] = wmma_bf16(af, bf, acc[j]);
            }
        }
#pragma unroll
        for (int j = 0; j < 4; ++j) {
            int col = n0 + j * 16 + (lane & 15);
            float bv = bias[col];
            int mofs = m0 + ((lane & 16) ? 8 : 0);
#pragma unroll
            for (int r = 0; r < 8; ++r) {
                int m = mofs + r;
                float val = acc[j][r] + bv;
                if (GELU) {
                    hout[m * D512 + col] = f2bf(gelu_exact(val));
                } else {
                    if (m < S25) xres[m * D512 + col] += val;
                }
            }
        }
    }
}

// ---------- weight conversion ----------
extern "C" __global__ void ne_cvt_bf16(const float* __restrict__ in, u16* __restrict__ out, int n) {
    int i = blockIdx.x * blockDim.x + threadIdx.x;
    int stride = gridDim.x * blockDim.x;
    for (; i < n; i += stride) out[i] = f2bf(in[i]);
}

// ---------- main fused transformer kernel: one workgroup per batch element ----------
extern "C" __global__ __launch_bounds__(256, 1)
void ne_main(const float* __restrict__ nf, const float* __restrict__ bias_table,
             const float* __restrict__ ln1_g, const float* __restrict__ ln1_b,
             const u16* __restrict__ wqkv_bf, const float* __restrict__ bqkv,
             const u16* __restrict__ wo_bf, const float* __restrict__ bo,
             const float* __restrict__ ln2_g, const float* __restrict__ ln2_b,
             const u16* __restrict__ w1_bf, const float* __restrict__ b1,
             const u16* __restrict__ w2_bf, const float* __restrict__ b2,
             const float* __restrict__ lnf_g, const float* __restrict__ lnf_b,
             const float* __restrict__ pred_w, const float* __restrict__ pred_b,
             float* __restrict__ out_x, float* __restrict__ out_pred) {
    extern __shared__ __align__(16) char smem[];
    float* xs  = (float*)smem;                                   // 25*512 f32   = 51200 B
    u16*   xn  = (u16*)(smem + 51200);                           // 32*512 bf16  = 32768 B (also o-buf / ffn in)
    u16*   qb  = (u16*)(smem + 51200 + 32768);                   // 16*32*32 bf16
    u16*   kb  = qb + NH * 32 * 32;                              // 16*32*32 bf16
    u16*   vT  = kb + NH * 32 * 32;                              // 16*32*32 bf16 (transposed per head)
    float* sc  = (float*)(smem + 51200 + 32768 + 3 * 32768);     // 16*32*32 f32 = 65536 B (reused as h_bf)
    u16*   at  = (u16*)((char*)sc + 65536);                      // 16*32*32 bf16
    float* pooled = (float*)((char*)at + 32768);                 // 512 f32
    // total = 282624 B < 320 KB WGP LDS

    int b = blockIdx.x;
    int tid = threadIdx.x, wave = tid >> 5, lane = tid & 31;

    // ---- stage x (25x512 f32) into LDS with CDNA5 async copy engine ----
    {
        const char* src = (const char*)(nf + (size_t)b * S25 * D512);
        u32 xs_base = (u32)(uintptr_t)xs;   // low 32 bits of generic shared ptr = LDS byte offset
        for (int i = tid; i < S25 * D512 * 4 / 16; i += 256)
            async_g2l_b128(xs_base + i * 16, src + i * 16);
        wait_async0();
    }
    __syncthreads();

    const float scale = 0.17677669529663687f; // 1/sqrt(32)

    for (int l = 0; l < 3; ++l) {
        const u16* Wqkv = wqkv_bf + (size_t)l * 1536 * D512;
        const float* bq = bqkv + l * 1536;
        __builtin_prefetch(Wqkv, 0, 1);

        // ---- LN1 -> xn (bf16, padded) ----
        ln_to_bf16(xs, ln1_g + l * D512, ln1_b + l * D512, xn);
        __syncthreads();

        // ---- QKV GEMM: [32x512] @ [1536x512]^T, scatter to q/k/vT head tiles ----
        for (int g = wave; g < 48; g += 8) {   // 2 m-tiles x 24 groups of 4 n-tiles (96 n-tiles)
            int mt = g / 24, ng = g % 24;
            int m0 = mt * 16, n0 = ng * 64;
            v8f acc[4] = { zero8(), zero8(), zero8(), zero8() };
            for (int kt = 0; kt < 16; ++kt) {
                v16bf af = load_a_frag(xn, D512, m0, kt * 32);
#pragma unroll
                for (int j = 0; j < 4; ++j) {
                    v16bf bf = load_b_frag(Wqkv, D512, n0 + j * 16, kt * 32);
                    acc[j] = wmma_bf16(af, bf, acc[j]);
                }
            }
#pragma unroll
            for (int j = 0; j < 4; ++j) {
                int col = n0 + j * 16 + (lane & 15);    // 0..1535
                int reg = col >> 9;                     // 0=q 1=k 2=v
                int within = col & 511;
                int head = within >> 5, d = within & 31;
                float bv = bq[col];
                int mofs = m0 + ((lane & 16) ? 8 : 0);
#pragma unroll
                for (int r = 0; r < 8; ++r) {
                    int m = mofs + r;
                    u16 h16 = f2bf(acc[j][r] + bv);
                    if (reg == 0)      qb[head * 1024 + m * 32 + d] = h16;
                    else if (reg == 1) kb[head * 1024 + m * 32 + d] = h16;
                    else               vT[head * 1024 + d * 32 + m] = h16;
                }
            }
        }
        __syncthreads();

        // ---- scores = q @ k^T (per head, single K=32 tile) ----
        for (int t = wave; t < 64; t += 8) {
            int h = t >> 2, mt = (t >> 1) & 1, nt = t & 1;
            v8f acc = zero8();
            v16bf af = load_a_frag(qb + h * 1024, 32, mt * 16, 0);
            v16bf bf = load_b_frag(kb + h * 1024, 32, nt * 16, 0);
            acc = wmma_bf16(af, bf, acc);
            int n = nt * 16 + (lane & 15);
            int mofs = mt * 16 + ((lane & 16) ? 8 : 0);
#pragma unroll
            for (int r = 0; r < 8; ++r) sc[h * 1024 + (mofs + r) * 32 + n] = acc[r];
        }
        __syncthreads();

        // ---- softmax with relative-position bias -> attn bf16 ----
        for (int t = tid; t < NH * S25; t += 256) {
            int h = t / S25, q = t % S25;
            int qx = q / 5, qy = q % 5;
            float vals[S25];
            float mx = -1e30f;
#pragma unroll
            for (int kk = 0; kk < S25; ++kk) {
                int kx = kk / 5, ky = kk % 5;
                int ridx = (qx - kx + 4) * 9 + (qy - ky + 4);
                float s = sc[h * 1024 + q * 32 + kk] * scale + bias_table[ridx * NH + h];
                vals[kk] = s;
                mx = fmaxf(mx, s);
            }
            float sum = 0.f;
#pragma unroll
            for (int kk = 0; kk < S25; ++kk) { float e = __expf(vals[kk] - mx); vals[kk] = e; sum += e; }
            float inv = 1.0f / sum;
#pragma unroll
            for (int kk = 0; kk < S25; ++kk) at[h * 1024 + q * 32 + kk] = f2bf(vals[kk] * inv);
#pragma unroll
            for (int kk = S25; kk < 32; ++kk) at[h * 1024 + q * 32 + kk] = 0;
        }
        for (int t = tid; t < NH * 7 * 32; t += 256) {   // zero pad rows q=25..31
            int h = t / (7 * 32), rr = t % (7 * 32);
            at[h * 1024 + (S25 + rr / 32) * 32 + (rr & 31)] = 0;
        }
        __syncthreads();

        // ---- o = attn @ v  -> o-buf (reuses xn) [32][512] bf16 ----
        for (int t = wave; t < 64; t += 8) {
            int h = t >> 2, mt = (t >> 1) & 1, nt = t & 1;
            v8f acc = zero8();
            v16bf af = load_a_frag(at + h * 1024, 32, mt * 16, 0);
            v16bf bf = load_b_frag(vT + h * 1024, 32, nt * 16, 0);
            acc = wmma_bf16(af, bf, acc);
            int d = nt * 16 + (lane & 15);
            int mofs = mt * 16 + ((lane & 16) ? 8 : 0);
#pragma unroll
            for (int r = 0; r < 8; ++r)
                xn[(mofs + r) * D512 + h * HD32 + d] = f2bf(acc[r]);
        }
        __syncthreads();

        // ---- x += o @ wo^T + bo ----
        gemm_ffn<false>(xn, wo_bf + (size_t)l * D512 * D512, bo + l * D512, xs, nullptr);
        __syncthreads();

        // ---- LN2 -> xn ----
        ln_to_bf16(xs, ln2_g + l * D512, ln2_b + l * D512, xn);
        __syncthreads();

        // ---- h = gelu(xn @ w1^T + b1) -> h_bf (reuses scores buffer) ----
        u16* hb = (u16*)sc;
        gemm_ffn<true>(xn, w1_bf + (size_t)l * D512 * D512, b1 + l * D512, nullptr, hb);
        __syncthreads();

        // ---- x += h @ w2^T + b2 ----
        gemm_ffn<false>(hb, w2_bf + (size_t)l * D512 * D512, b2 + l * D512, xs, nullptr);
        __syncthreads();
    }

    // ---- final LN -> out_x, accumulate pooled mean ----
    for (int i = tid; i < D512; i += 256) pooled[i] = 0.f;
    __syncthreads();
    {
        float* ox = out_x + (size_t)b * S25 * D512;
        for (int t = wave; t < S25; t += 8) {
            const float4* rp = (const float4*)(xs + t * D512 + lane * 16);
            float4 v[4];
            float s = 0.f, ss = 0.f;
#pragma unroll
            for (int i = 0; i < 4; ++i) {
                v[i] = rp[i];
                s  += v[i].x + v[i].y + v[i].z + v[i].w;
                ss += v[i].x*v[i].x + v[i].y*v[i].y + v[i].z*v[i].z + v[i].w*v[i].w;
            }
#pragma unroll
            for (int off = 16; off; off >>= 1) {
                s  += __shfl_xor(s, off, 32);
                ss += __shfl_xor(ss, off, 32);
            }
            float mu  = s  * (1.f / 512.f);
            float var = ss * (1.f / 512.f) - mu * mu;
            float r   = rsqrtf(var + 1e-5f);
#pragma unroll
            for (int i = 0; i < 4; ++i) {
                float vv[4] = { v[i].x, v[i].y, v[i].z, v[i].w };
#pragma unroll
                for (int j = 0; j < 4; ++j) {
                    int col = lane * 16 + i * 4 + j;
                    float val = (vv[j] - mu) * r * lnf_g[col] + lnf_b[col];
                    ox[t * D512 + col] = val;
                    atomicAdd(&pooled[col], val * (1.f / 25.f));
                }
            }
        }
    }
    __syncthreads();

    // ---- pred = pooled @ pred_w^T + pred_b ----
    for (int g = tid; g < 250; g += 256) {
        const float* wr = pred_w + g * D512;
        float acc = 0.f;
        for (int i = 0; i < D512; i += 4) {
            acc += pooled[i]     * wr[i]
                 + pooled[i + 1] * wr[i + 1]
                 + pooled[i + 2] * wr[i + 2]
                 + pooled[i + 3] * wr[i + 3];
        }
        out_pred[(size_t)b * 250 + g] = acc + pred_b[g];
    }
}

extern "C" void kernel_launch(void* const* d_in, const int* in_sizes, int n_in,
                              void* d_out, int out_size, void* d_ws, size_t ws_size,
                              hipStream_t stream) {
    const float* nf        = (const float*)d_in[0];
    const float* bias_tab  = (const float*)d_in[1];
    const float* ln1_g     = (const float*)d_in[2];
    const float* ln1_b     = (const float*)d_in[3];
    const float* wqkv      = (const float*)d_in[4];
    const float* bqkv      = (const float*)d_in[5];
    const float* wo        = (const float*)d_in[6];
    const float* bo        = (const float*)d_in[7];
    const float* ln2_g     = (const float*)d_in[8];
    const float* ln2_b     = (const float*)d_in[9];
    const float* w1        = (const float*)d_in[10];
    const float* b1        = (const float*)d_in[11];
    const float* w2        = (const float*)d_in[12];
    const float* b2        = (const float*)d_in[13];
    const float* lnf_g     = (const float*)d_in[14];
    const float* lnf_b     = (const float*)d_in[15];
    const float* pred_w    = (const float*)d_in[16];
    const float* pred_b    = (const float*)d_in[17];

    int B = in_sizes[0] / (S25 * D512);

    // bf16 weight shadows in workspace (L2-resident, ~9.4 MB)
    u16* ws       = (u16*)d_ws;
    u16* wqkv_bf  = ws;
    u16* wo_bf    = wqkv_bf + (size_t)3 * 1536 * 512;
    u16* w1_bf    = wo_bf   + (size_t)3 * 512 * 512;
    u16* w2_bf    = w1_bf   + (size_t)3 * 512 * 512;

    ne_cvt_bf16<<<2048, 256, 0, stream>>>(wqkv, wqkv_bf, 3 * 1536 * 512);
    ne_cvt_bf16<<<1024, 256, 0, stream>>>(wo,   wo_bf,   3 * 512 * 512);
    ne_cvt_bf16<<<1024, 256, 0, stream>>>(w1,   w1_bf,   3 * 512 * 512);
    ne_cvt_bf16<<<1024, 256, 0, stream>>>(w2,   w2_bf,   3 * 512 * 512);

    float* out_x    = (float*)d_out;
    float* out_pred = out_x + (size_t)B * S25 * D512;

    size_t smem_bytes = 51200 + 32768 + 3 * 32768 + 65536 + 32768 + 2048; // 282624
    hipFuncSetAttribute((const void*)ne_main, hipFuncAttributeMaxDynamicSharedMemorySize,
                        (int)smem_bytes);

    ne_main<<<B, 256, smem_bytes, stream>>>(
        nf, bias_tab, ln1_g, ln1_b, wqkv_bf, bqkv, wo_bf, bo,
        ln2_g, ln2_b, w1_bf, b1, w2_bf, b2, lnf_g, lnf_b,
        pred_w, pred_b, out_x, out_pred);
}
